// CTCTransformerModel_7241314861805
// MI455X (gfx1250) — matrix operand order repaired and checked by reading, not compiled, not run on hardware
//
#include <hip/hip_runtime.h>
#include <stdint.h>

// ---------------------------------------------------------------------------
// CDNA5 (gfx1250) WMMA types
// ---------------------------------------------------------------------------
typedef __attribute__((ext_vector_type(16))) __bf16 v16bf;
typedef __attribute__((ext_vector_type(8)))  float  v8f;

union Frag16 { uint4 q[2]; v16bf v; };

// ---------------------------------------------------------------------------
// helpers
// ---------------------------------------------------------------------------
__device__ __forceinline__ uint16_t f2bf(float f) {
    uint32_t u = __builtin_bit_cast(uint32_t, f);
    uint32_t r = (u + 0x7FFFu + ((u >> 16) & 1u)) >> 16;
    return (uint16_t)r;
}

__device__ __forceinline__ float gelu_erf(float x) {
    return 0.5f * x * (1.0f + erff(x * 0.70710678118654752f));
}

// gfx1250 async global->LDS copy (16B per lane), tracked by ASYNCcnt.
__device__ __forceinline__ void async_b128(uint32_t ldsOff, const uint16_t* g) {
    asm volatile("global_load_async_to_lds_b128 %0, %1, off"
                 :: "v"(ldsOff), "v"(g) : "memory");
}
__device__ __forceinline__ void wait_async0() {
    asm volatile("s_wait_asynccnt 0" ::: "memory");
}

// ---------------------------------------------------------------------------
// Direct bf16 WMMA GEMM (fallback for shapes not tiling to 64x128).
// A [M,K] bf16 row-major, Bt [N,K] bf16 row-major (B transposed), C [M,N] fp32.
// Wave computes 32x32 C tile: 4x v_wmma_f32_16x16x32_bf16 per 32-wide K step.
// M,N multiples of 32; K multiple of 32.  grid.y = batch (strided).
// ---------------------------------------------------------------------------
__global__ __launch_bounds__(256)
void k_gemm_bf16(const uint16_t* __restrict__ A,
                 const uint16_t* __restrict__ Bt,
                 float* __restrict__ C,
                 const float* __restrict__ bias, int Nbias,
                 int M, int N, int K,
                 long long sA, long long sB, long long sC)
{
    const int lane = threadIdx.x & 31;
    const int wave = threadIdx.x >> 5;
    const int tilesN = N >> 5;
    const int totalTiles = (M >> 5) * tilesN;
    const int tile = blockIdx.x * 8 + wave;
    if (tile >= totalTiles) return;
    const int tm = (tile / tilesN) << 5;
    const int tn = (tile % tilesN) << 5;

    const uint16_t* Ab = A  + (long long)blockIdx.y * sA;
    const uint16_t* Bb = Bt + (long long)blockIdx.y * sB;
    float*          Cb = C  + (long long)blockIdx.y * sC;

    v8f acc00 = {}, acc01 = {}, acc10 = {}, acc11 = {};

    const int l15 = lane & 15;
    const int aK  = (lane < 16) ? 0 : 8;    // A per-lane K chunks: aK, aK+16
    const int bK  = (lane < 16) ? 0 : 8;    // B per-lane K chunks: bK, bK+16

    const uint16_t* aRow0 = Ab + (size_t)(tm + l15) * K;
    const uint16_t* aRow1 = Ab + (size_t)(tm + 16 + l15) * K;
    const uint16_t* bRow0 = Bb + (size_t)(tn + l15) * K;
    const uint16_t* bRow1 = Bb + (size_t)(tn + 16 + l15) * K;

    for (int kk = 0; kk < K; kk += 32) {
        Frag16 a0, a1, b0, b1;
        a0.q[0] = *(const uint4*)(aRow0 + kk + aK);
        a0.q[1] = *(const uint4*)(aRow0 + kk + aK + 16);
        a1.q[0] = *(const uint4*)(aRow1 + kk + aK);
        a1.q[1] = *(const uint4*)(aRow1 + kk + aK + 16);
        b0.q[0] = *(const uint4*)(bRow0 + kk + bK);
        b0.q[1] = *(const uint4*)(bRow0 + kk + bK + 16);
        b1.q[0] = *(const uint4*)(bRow1 + kk + bK);
        b1.q[1] = *(const uint4*)(bRow1 + kk + bK + 16);
        if (kk + 32 < K) {   // gfx1250 global_prefetch_b8 to hide HBM latency
            __builtin_prefetch(aRow0 + kk + 32, 0, 1);
            __builtin_prefetch(bRow0 + kk + 32, 0, 1);
        }
        acc00 = __builtin_amdgcn_wmma_f32_16x16x32_bf16(false, a0.v, false, b0.v, (short)0, acc00, false, false);
        acc01 = __builtin_amdgcn_wmma_f32_16x16x32_bf16(false, a0.v, false, b1.v, (short)0, acc01, false, false);
        acc10 = __builtin_amdgcn_wmma_f32_16x16x32_bf16(false, a1.v, false, b0.v, (short)0, acc10, false, false);
        acc11 = __builtin_amdgcn_wmma_f32_16x16x32_bf16(false, a1.v, false, b1.v, (short)0, acc11, false, false);
    }

    const int rowOff = (lane < 16) ? 0 : 8;
    float bias0 = 0.f, bias1 = 0.f;
    if (bias) {
        int c0 = tn + l15, c1 = tn + 16 + l15;
        bias0 = (c0 < Nbias) ? bias[c0] : 0.f;
        bias1 = (c1 < Nbias) ? bias[c1] : 0.f;
    }
#pragma unroll
    for (int r = 0; r < 8; ++r) {
        int row0 = tm + rowOff + r;
        int row1 = tm + 16 + rowOff + r;
        Cb[(size_t)row0 * N + tn + l15]      = acc00[r] + bias0;
        Cb[(size_t)row0 * N + tn + 16 + l15] = acc01[r] + bias1;
        Cb[(size_t)row1 * N + tn + l15]      = acc10[r] + bias0;
        Cb[(size_t)row1 * N + tn + 16 + l15] = acc11[r] + bias1;
    }
}

// ---------------------------------------------------------------------------
// LDS-staged bf16 WMMA GEMM (requires M%64==0, N%128==0).
// Block = 8 waves -> 64x128 C macro-tile (waves arranged 2(M) x 4(N)).
// Per 32-wide K step: A(64x32) + B(128x32) bf16 tiles (12 KB) are staged into
// double-buffered LDS with global_load_async_to_lds_b128 (ASYNCcnt), then
// each wave reads fragments via ds_load_b128 and issues 4 WMMAs.
// Async copy of tile k+1 overlaps WMMA on tile k.
// ---------------------------------------------------------------------------
__global__ __launch_bounds__(256)
void k_gemm_bf16_lds(const uint16_t* __restrict__ A,
                     const uint16_t* __restrict__ Bt,
                     float* __restrict__ C,
                     const float* __restrict__ bias, int Nbias,
                     int M, int N, int K,
                     long long sA, long long sB, long long sC)
{
    __shared__ __align__(16) uint16_t smem[2 * 6144];  // 2 x (A 64x32 + B 128x32)
    const int tid  = threadIdx.x;
    const int lane = tid & 31;
    const int wave = tid >> 5;
    const int mi = wave >> 2;         // 0..1 : M sub-tile
    const int ni = wave & 3;          // 0..3 : N sub-tile
    const int tilesNB = N >> 7;
    const int tm = (blockIdx.x / tilesNB) << 6;
    const int tn = (blockIdx.x % tilesNB) << 7;

    const uint16_t* Ab = A  + (long long)blockIdx.y * sA;
    const uint16_t* Bb = Bt + (long long)blockIdx.y * sB;
    float*          Cb = C  + (long long)blockIdx.y * sC;

    // per-thread async-copy assignments (16B chunks)
    const int arow = tid >> 2, ach = tid & 3;          // A: 64 rows x 4 chunks
    const int brow0 = tid >> 2;                        // B: 128 rows x 4 chunks
    const int brow1 = 64 + (tid >> 2);
    const int bch = tid & 3;
    const uint16_t* gA  = Ab + (size_t)(tm + arow)  * K + ach * 8;
    const uint16_t* gB0 = Bb + (size_t)(tn + brow0) * K + bch * 8;
    const uint16_t* gB1 = Bb + (size_t)(tn + brow1) * K + bch * 8;

    const uint32_t ldsBase = (uint32_t)(uintptr_t)(&smem[0]);
    const uint32_t oA  = (uint32_t)(arow  * 64 + ach * 16);
    const uint32_t oB0 = (uint32_t)(4096 + brow0 * 64 + bch * 16);
    const uint32_t oB1 = (uint32_t)(4096 + brow1 * 64 + bch * 16);

    v8f acc00 = {}, acc01 = {}, acc10 = {}, acc11 = {};

    const int l15 = lane & 15;
    const int aK  = (lane < 16) ? 0 : 8;
    const int bK  = (lane < 16) ? 0 : 8;
    // fragment element offsets (uint16 units) within one LDS buffer
    const int aIdx0 = (mi * 32 + l15) * 32;
    const int aIdx1 = (mi * 32 + 16 + l15) * 32;
    const int bIdx0 = 2048 + (ni * 32 + l15) * 32;
    const int bIdx1 = 2048 + (ni * 32 + 16 + l15) * 32;

    // prologue: stage first K chunk into buffer 0
    {
        uint32_t bb = ldsBase;
        async_b128(bb + oA,  gA);
        async_b128(bb + oB0, gB0);
        async_b128(bb + oB1, gB1);
    }
    int buf = 0;
    for (int kk = 0; kk < K; kk += 32) {
        wait_async0();          // my wave's staged data is in LDS
        __syncthreads();        // everyone's data is in LDS; prior reads done
        if (kk + 32 < K) {      // overlap: stage next chunk into other buffer
            uint32_t bb = ldsBase + (uint32_t)(buf ^ 1) * 12288u;
            async_b128(bb + oA,  gA  + kk + 32);
            async_b128(bb + oB0, gB0 + kk + 32);
            async_b128(bb + oB1, gB1 + kk + 32);
        }
        const uint16_t* s = smem + buf * 6144;
        Frag16 a0, a1, b0, b1;
        a0.q[0] = *(const uint4*)(s + aIdx0 + aK);
        a0.q[1] = *(const uint4*)(s + aIdx0 + aK + 16);
        a1.q[0] = *(const uint4*)(s + aIdx1 + aK);
        a1.q[1] = *(const uint4*)(s + aIdx1 + aK + 16);
        b0.q[0] = *(const uint4*)(s + bIdx0 + bK);
        b0.q[1] = *(const uint4*)(s + bIdx0 + bK + 16);
        b1.q[0] = *(const uint4*)(s + bIdx1 + bK);
        b1.q[1] = *(const uint4*)(s + bIdx1 + bK + 16);
        acc00 = __builtin_amdgcn_wmma_f32_16x16x32_bf16(false, a0.v, false, b0.v, (short)0, acc00, false, false);
        acc01 = __builtin_amdgcn_wmma_f32_16x16x32_bf16(false, a0.v, false, b1.v, (short)0, acc01, false, false);
        acc10 = __builtin_amdgcn_wmma_f32_16x16x32_bf16(false, a1.v, false, b0.v, (short)0, acc10, false, false);
        acc11 = __builtin_amdgcn_wmma_f32_16x16x32_bf16(false, a1.v, false, b1.v, (short)0, acc11, false, false);
        __syncthreads();        // all waves done reading buf before overwrite
        buf ^= 1;
    }

    const int wm = tm + mi * 32;
    const int wn = tn + ni * 32;
    const int rowOff = (lane < 16) ? 0 : 8;
    float bias0 = 0.f, bias1 = 0.f;
    if (bias) {
        int c0 = wn + l15, c1 = wn + 16 + l15;
        bias0 = (c0 < Nbias) ? bias[c0] : 0.f;
        bias1 = (c1 < Nbias) ? bias[c1] : 0.f;
    }
#pragma unroll
    for (int r = 0; r < 8; ++r) {
        int row0 = wm + rowOff + r;
        int row1 = wm + 16 + rowOff + r;
        Cb[(size_t)row0 * N + wn + l15]      = acc00[r] + bias0;
        Cb[(size_t)row0 * N + wn + 16 + l15] = acc01[r] + bias1;
        Cb[(size_t)row1 * N + wn + l15]      = acc10[r] + bias0;
        Cb[(size_t)row1 * N + wn + 16 + l15] = acc11[r] + bias1;
    }
}

// ---------------------------------------------------------------------------
// data movement / elementwise kernels
// ---------------------------------------------------------------------------

// chunk (7 overlapping windows of 320, stride 256) + space-to-depth(4), NHWC out
__global__ void k_s2d(const float* __restrict__ x, float* __restrict__ c) {
    int idx = blockIdx.x * blockDim.x + threadIdx.x;
    if (idx >= 28 * 800 * 16) return;
    int ch = idx & 15;
    int t  = idx >> 4;
    int w4 = t % 80; t /= 80;
    int h4 = t % 10; t /= 10;
    int bn = t;
    int b = bn / 7, n = bn % 7;
    int a = ch >> 2, bc = ch & 3;
    int hh = h4 * 4 + a;
    int ww = n * 256 + w4 * 4 + bc;
    c[idx] = x[((size_t)b * 40 + hh) * 1856 + ww];
}

// im2col for 3x3 pad1 on [BN,10,80,Cin] NHWC -> bf16 [22400, Kp]
__global__ void k_im2col3(const float* __restrict__ src, uint16_t* __restrict__ dst,
                          int Cin, int Kp) {
    long long idx = (long long)blockIdx.x * blockDim.x + threadIdx.x;
    long long total = 22400LL * Kp;
    if (idx >= total) return;
    int col = (int)(idx % Kp);
    int m   = (int)(idx / Kp);
    uint16_t o = 0;
    if (col < Cin * 9) {
        int ci = col / 9, r = col % 9, ky = r / 3, kx = r % 3;
        int bn = m / 800, sp = m % 800, y = sp / 80, xw = sp % 80;
        int iy = y + ky - 1, ix = xw + kx - 1;
        if (iy >= 0 && iy < 10 && ix >= 0 && ix < 80)
            o = f2bf(src[(((size_t)bn * 10 + iy) * 80 + ix) * Cin + ci]);
    }
    dst[idx] = o;
}

// im2col for reduce conv (10,3) pad (0,1): [BN,10,80,64] -> bf16 [2240,1920]
__global__ void k_im2col_red(const float* __restrict__ src, uint16_t* __restrict__ dst) {
    long long idx = (long long)blockIdx.x * blockDim.x + threadIdx.x;
    if (idx >= 2240LL * 1920) return;
    int col = (int)(idx % 1920);
    int m   = (int)(idx / 1920);
    int ci = col / 30, r = col % 30, ky = r / 3, kx = r % 3;
    int bn = m / 80, xw = m % 80;
    int ix = xw + kx - 1;
    uint16_t o = 0;
    if (ix >= 0 && ix < 80)
        o = f2bf(src[(((size_t)bn * 10 + ky) * 80 + ix) * 64 + ci]);
    dst[idx] = o;
}

__global__ void k_bn_act(const float* __restrict__ in, float* __restrict__ out,
                         const float* __restrict__ s, const float* __restrict__ b,
                         long long total, int N, int act) {
    long long idx = (long long)blockIdx.x * blockDim.x + threadIdx.x;
    if (idx >= total) return;
    int ch = (int)(idx % N);
    float v = in[idx] * s[ch] + b[ch];
    if (act) v = gelu_erf(v);
    out[idx] = v;
}

__global__ void k_res_gelu(float* __restrict__ y, const float* __restrict__ r,
                           long long total) {
    long long idx = (long long)blockIdx.x * blockDim.x + threadIdx.x;
    if (idx >= total) return;
    y[idx] = gelu_erf(r[idx] + y[idx]);
}

__global__ void k_gelu_ip(float* __restrict__ y, long long total) {
    long long idx = (long long)blockIdx.x * blockDim.x + threadIdx.x;
    if (idx >= total) return;
    y[idx] = gelu_erf(y[idx]);
}

__global__ void k_convert(const float* __restrict__ src, uint16_t* __restrict__ dst,
                          long long total) {
    long long idx = (long long)blockIdx.x * blockDim.x + threadIdx.x;
    if (idx >= total) return;
    dst[idx] = f2bf(src[idx]);
}

// src [N,K] -> dst [N,Kp] bf16, zero-padded K
__global__ void k_convert_pad(const float* __restrict__ src, uint16_t* __restrict__ dst,
                              int N, int K, int Kp) {
    long long idx = (long long)blockIdx.x * blockDim.x + threadIdx.x;
    if (idx >= (long long)N * Kp) return;
    int k = (int)(idx % Kp);
    int n = (int)(idx / Kp);
    dst[idx] = (k < K) ? f2bf(src[(size_t)n * K + k]) : (uint16_t)0;
}

// batched: src [batch,K,N] -> dst [batch,Np,K] bf16 (transpose, zero-pad N->Np)
__global__ void k_convert_transB(const float* __restrict__ src, uint16_t* __restrict__ dst,
                                 int K, int N, int Np, int batch) {
    long long idx = (long long)blockIdx.x * blockDim.x + threadIdx.x;
    long long per = (long long)Np * K;
    if (idx >= per * batch) return;
    int l  = (int)(idx / per);
    long long r = idx % per;
    int nn = (int)(r / K);
    int kk = (int)(r % K);
    dst[idx] = (nn < N) ? f2bf(src[(size_t)l * K * N + (size_t)kk * N + nn]) : (uint16_t)0;
}

// stitch halo-trimmed chunks + sinusoidal PE -> h [4,464,256]
__global__ void k_stitch_pe(const float* __restrict__ f, float* __restrict__ h) {
    int idx = blockIdx.x * blockDim.x + threadIdx.x;
    if (idx >= 4 * 464 * 256) return;
    int cch = idx & 255;
    int t   = (idx >> 8) % 464;
    int b   = idx / (464 * 256);
    int n, xw;
    if (t < 72)        { n = 0; xw = t; }
    else if (t >= 392) { n = 6; xw = t - 392 + 8; }
    else               { n = 1 + (t - 72) / 64; xw = 8 + (t - 72) % 64; }
    int bn = b * 7 + n;
    int i2 = cch >> 1;
    float div = __expf(-9.2103403719761836f * (float)(2 * i2) / 256.0f);
    float ang = (float)t * div;
    float pe = (cch & 1) ? __cosf(ang) : __sinf(ang);
    h[idx] = f[((size_t)bn * 80 + xw) * 256 + cch] + pe;
}

// split qkv [B*464,768] -> Qb,Kb bf16 [16,480,64] (zero-padded rows), Vt [16,64,480]
__global__ void k_qkv_split(const float* __restrict__ qkv,
                            uint16_t* __restrict__ Qb, uint16_t* __restrict__ Kb,
                            uint16_t* __restrict__ Vt) {
    int idx = blockIdx.x * blockDim.x + threadIdx.x;
    if (idx >= 16 * 480 * 64) return;
    int d  = idx & 63;
    int t  = (idx >> 6) % 480;
    int bh = idx / (480 * 64);
    int b = bh >> 2, h = bh & 3;
    uint16_t qv = 0, kv = 0, vv = 0;
    if (t < 464) {
        const float* p = qkv + (size_t)(b * 464 + t) * 768 + h * 64 + d;
        qv = f2bf(p[0]);
        kv = f2bf(p[256]);
        vv = f2bf(p[512]);
    }
    Qb[idx] = qv;
    Kb[idx] = kv;
    Vt[((size_t)bh * 64 + d) * 480 + t] = vv;
}

// softmax over scores [16,480,480] with rel-pos bias, writes bf16 attn (padded)
__global__ __launch_bounds__(256)
void k_softmax_bias(const float* __restrict__ sc, const float* __restrict__ rel,
                    uint16_t* __restrict__ attn) {
    __shared__ float red[256];
    const int q   = blockIdx.x;   // 0..479
    const int bh  = blockIdx.y;   // 0..15
    const int h   = bh & 3;
    const int tid = threadIdx.x;
    const float* row  = sc   + ((size_t)bh * 480 + q) * 480;
    uint16_t*    arow = attn + ((size_t)bh * 480 + q) * 480;
    if (q >= 464) {
        for (int k = tid; k < 480; k += 256) arow[k] = 0;
        return;
    }
    float lmax = -3.0e38f;
    for (int k = tid; k < 464; k += 256) {
        float v = row[k] * 0.125f + rel[(size_t)(q - k + 512) * 4 + h];
        lmax = fmaxf(lmax, v);
    }
    red[tid] = lmax; __syncthreads();
    for (int s = 128; s > 0; s >>= 1) { if (tid < s) red[tid] = fmaxf(red[tid], red[tid + s]); __syncthreads(); }
    float m = red[0]; __syncthreads();
    float lsum = 0.f;
    for (int k = tid; k < 464; k += 256) {
        float v = row[k] * 0.125f + rel[(size_t)(q - k + 512) * 4 + h];
        lsum += __expf(v - m);
    }
    red[tid] = lsum; __syncthreads();
    for (int s = 128; s > 0; s >>= 1) { if (tid < s) red[tid] += red[tid + s]; __syncthreads(); }
    float inv = 1.f / red[0];
    for (int k = tid; k < 480; k += 256) {
        uint16_t o = 0;
        if (k < 464) {
            float v = row[k] * 0.125f + rel[(size_t)(q - k + 512) * 4 + h];
            o = f2bf(__expf(v - m) * inv);
        }
        arow[k] = o;
    }
}

// merge heads: o [16,480,64] -> ctx bf16 [1856,256]
__global__ void k_merge_ctx(const float* __restrict__ o, uint16_t* __restrict__ ctx) {
    int idx = blockIdx.x * blockDim.x + threadIdx.x;
    if (idx >= 4 * 464 * 256) return;
    int col = idx & 255;
    int t   = (idx >> 8) % 464;
    int b   = idx / (464 * 256);
    int h = col >> 6, d = col & 63;
    ctx[idx] = f2bf(o[(((size_t)(b * 4 + h)) * 480 + t) * 64 + d]);
}

// h = LayerNorm(h + delta) * g + b, rows of 256
__global__ __launch_bounds__(256)
void k_add_ln(float* __restrict__ h, const float* __restrict__ d,
              const float* __restrict__ g, const float* __restrict__ b) {
    __shared__ float red[256];
    const int row = blockIdx.x;
    const int tid = threadIdx.x;
    float v = h[(size_t)row * 256 + tid] + d[(size_t)row * 256 + tid];
    red[tid] = v; __syncthreads();
    for (int s = 128; s > 0; s >>= 1) { if (tid < s) red[tid] += red[tid + s]; __syncthreads(); }
    float mu = red[0] * (1.f / 256.f); __syncthreads();
    float c = v - mu;
    red[tid] = c * c; __syncthreads();
    for (int s = 128; s > 0; s >>= 1) { if (tid < s) red[tid] += red[tid + s]; __syncthreads(); }
    float var = red[0] * (1.f / 256.f);
    h[(size_t)row * 256 + tid] = c * rsqrtf(var + 1e-5f) * g[tid] + b[tid];
}

// extract logits [1856,128] -> d_out [1856,100]
__global__ void k_extract(const float* __restrict__ C, float* __restrict__ out) {
    int idx = blockIdx.x * blockDim.x + threadIdx.x;
    if (idx >= 1856 * 100) return;
    int m = idx / 100, v = idx % 100;
    out[idx] = C[(size_t)m * 128 + v];
}

// ---------------------------------------------------------------------------
// host orchestration
// ---------------------------------------------------------------------------
static inline int cdiv(long long a, int b) { return (int)((a + b - 1) / b); }

static void gemm(hipStream_t st, const uint16_t* A, const uint16_t* Bt, float* C,
                 const float* bias, int Nbias, int M, int N, int K,
                 int batch, long long sA, long long sB, long long sC) {
    if ((M & 63) == 0 && (N & 127) == 0) {
        dim3 grid((M >> 6) * (N >> 7), batch, 1);
        k_gemm_bf16_lds<<<grid, 256, 0, st>>>(A, Bt, C, bias, Nbias, M, N, K, sA, sB, sC);
    } else {
        int tiles = (M / 32) * (N / 32);
        dim3 grid(cdiv(tiles, 8), batch, 1);
        k_gemm_bf16<<<grid, 256, 0, st>>>(A, Bt, C, bias, Nbias, M, N, K, sA, sB, sC);
    }
}

extern "C" void kernel_launch(void* const* d_in, const int* in_sizes, int n_in,
                              void* d_out, int out_size, void* d_ws, size_t ws_size,
                              hipStream_t stream) {
    (void)in_sizes; (void)n_in; (void)out_size; (void)ws_size;
    const float* x        = (const float*)d_in[0];
    const float* ci1_w    = (const float*)d_in[1];
    const float* bi1_s    = (const float*)d_in[2];
    const float* bi1_b    = (const float*)d_in[3];
    const float* ci2_w    = (const float*)d_in[4];
    const float* bi2_s    = (const float*)d_in[5];
    const float* bi2_b    = (const float*)d_in[6];
    const float* bc1_w    = (const float*)d_in[7];
    const float* bb1_s    = (const float*)d_in[8];
    const float* bb1_b    = (const float*)d_in[9];
    const float* bc2_w    = (const float*)d_in[10];
    const float* bb2_s    = (const float*)d_in[11];
    const float* bb2_b    = (const float*)d_in[12];
    const float* red_w    = (const float*)d_in[13];
    const float* red_s    = (const float*)d_in[14];
    const float* red_b    = (const float*)d_in[15];
    const float* qkv_w    = (const float*)d_in[16];
    const float* qkv_b    = (const float*)d_in[17];
    const float* out_w    = (const float*)d_in[18];
    const float* out_b    = (const float*)d_in[19];
    const float* ln1_g    = (const float*)d_in[20];
    const float* ln1_b    = (const float*)d_in[21];
    const float* ln2_g    = (const float*)d_in[22];
    const float* ln2_b    = (const float*)d_in[23];
    const float* mlp_w1   = (const float*)d_in[24];
    const float* mlp_b1   = (const float*)d_in[25];
    const float* mlp_w2   = (const float*)d_in[26];
    const float* mlp_b2   = (const float*)d_in[27];
    const float* rel_emb  = (const float*)d_in[28];
    const float* cls_w    = (const float*)d_in[29];
    const float* cls_b    = (const float*)d_in[30];
    float* out = (float*)d_out;

    // ---- workspace arena (deterministic fixed layout) ----
    char* ws = (char*)d_ws;
    size_t off = 0;
    auto take = [&](size_t bytes) -> char* {
        char* p = ws + off;
        off = (off + bytes + 255) & ~(size_t)255;
        return p;
    };
    uint16_t* wInit1 = (uint16_t*)take((size_t)128 * 160 * 2);
    uint16_t* wInit2 = (uint16_t*)take((size_t)64 * 128 * 2);
    uint16_t* wBlk1  = (uint16_t*)take((size_t)10 * 512 * 576 * 2);
    uint16_t* wBlk2  = (uint16_t*)take((size_t)10 * 64 * 512 * 2);
    uint16_t* wRed   = (uint16_t*)take((size_t)256 * 1920 * 2);
    uint16_t* wQkv   = (uint16_t*)take((size_t)16 * 768 * 256 * 2);
    uint16_t* wOutW  = (uint16_t*)take((size_t)16 * 256 * 256 * 2);
    uint16_t* wM1    = (uint16_t*)take((size_t)16 * 1024 * 256 * 2);
    uint16_t* wM2    = (uint16_t*)take((size_t)16 * 256 * 1024 * 2);
    uint16_t* wCls   = (uint16_t*)take((size_t)128 * 256 * 2);
    float*    s2d    = (float*)take((size_t)22400 * 16 * 4);
    uint16_t* imcol  = (uint16_t*)take((size_t)22400 * 576 * 2);  // also attn bf16
    float*    gemmC  = (float*)take((size_t)22400 * 512 * 4);     // also scores/qkv/mlp-hidden
    uint16_t* actb   = (uint16_t*)take((size_t)22400 * 512 * 2);  // bf16 activations
    float*    yres   = (float*)take((size_t)22400 * 64 * 4);
    float*    hbuf   = (float*)take((size_t)1856 * 256 * 4);
    uint16_t* Qb     = (uint16_t*)take((size_t)16 * 480 * 64 * 2);
    uint16_t* Kb     = (uint16_t*)take((size_t)16 * 480 * 64 * 2);
    uint16_t* Vt     = (uint16_t*)take((size_t)16 * 480 * 64 * 2);
    float*    oav    = (float*)take((size_t)16 * 480 * 64 * 4);
    float*    proj   = (float*)take((size_t)1856 * 256 * 4);

#define EL(total) <<<cdiv((long long)(total), 256), 256, 0, stream>>>

    // ---- weight conversion to bf16 [N,K] (transposed-B) layouts ----
    k_convert_pad   EL(128LL * 160)(ci1_w, wInit1, 128, 144, 160);
    k_convert       EL(64LL * 128)(ci2_w, wInit2, 64LL * 128);
    k_convert       EL(10LL * 512 * 576)(bc1_w, wBlk1, 10LL * 512 * 576);
    k_convert       EL(10LL * 64 * 512)(bc2_w, wBlk2, 10LL * 64 * 512);
    k_convert       EL(256LL * 1920)(red_w, wRed, 256LL * 1920);
    k_convert_transB EL(16LL * 768 * 256)(qkv_w, wQkv, 256, 768, 768, 16);
    k_convert_transB EL(16LL * 256 * 256)(out_w, wOutW, 256, 256, 256, 16);
    k_convert_transB EL(16LL * 1024 * 256)(mlp_w1, wM1, 256, 1024, 1024, 16);
    k_convert_transB EL(16LL * 256 * 1024)(mlp_w2, wM2, 1024, 256, 256, 16);
    k_convert_transB EL(128LL * 256)(cls_w, wCls, 256, 100, 128, 1);

    // ---- backbone ----
    k_s2d EL(28 * 800 * 16)(x, s2d);
    // init conv1 3x3: K padded 144->160
    k_im2col3 EL(22400LL * 160)(s2d, imcol, 16, 160);
    gemm(stream, imcol, wInit1, gemmC, nullptr, 0, 22400, 128, 160, 1, 0, 0, 0);
    k_bn_act EL(22400LL * 128)(gemmC, gemmC, bi1_s, bi1_b, 22400LL * 128, 128, 1);
    k_convert EL(22400LL * 128)(gemmC, actb, 22400LL * 128);
    // init conv2 1x1
    gemm(stream, actb, wInit2, gemmC, nullptr, 0, 22400, 64, 128, 1, 0, 0, 0);
    k_bn_act EL(22400LL * 64)(gemmC, yres, bi2_s, bi2_b, 22400LL * 64, 64, 1);

    // 10 fused inverted bottlenecks
    for (int i = 0; i < 10; ++i) {
        k_im2col3 EL(22400LL * 576)(yres, imcol, 64, 576);
        gemm(stream, imcol, wBlk1 + (size_t)i * 512 * 576, gemmC, nullptr, 0,
             22400, 512, 576, 1, 0, 0, 0);
        k_bn_act EL(22400LL * 512)(gemmC, gemmC, bb1_s + i * 512, bb1_b + i * 512,
                                   22400LL * 512, 512, 1);
        k_convert EL(22400LL * 512)(gemmC, actb, 22400LL * 512);
        gemm(stream, actb, wBlk2 + (size_t)i * 64 * 512, gemmC, nullptr, 0,
             22400, 64, 512, 1, 0, 0, 0);
        k_bn_act EL(22400LL * 64)(gemmC, gemmC, bb2_s + i * 64, bb2_b + i * 64,
                                  22400LL * 64, 64, 0);
        k_res_gelu EL(22400LL * 64)(yres, gemmC, 22400LL * 64);
    }

    // reduce conv (10,3) -> [2240,256], then stitch + PE -> h [4,464,256]
    k_im2col_red EL(2240LL * 1920)(yres, imcol);
    gemm(stream, imcol, wRed, gemmC, nullptr, 0, 2240, 256, 1920, 1, 0, 0, 0);
    k_bn_act EL(2240LL * 256)(gemmC, gemmC, red_s, red_b, 2240LL * 256, 256, 1);
    k_stitch_pe EL(4 * 464 * 256)(gemmC, hbuf);

    // ---- transformer encoder (16 layers) ----
    for (int i = 0; i < 16; ++i) {
        k_convert EL(1856LL * 256)(hbuf, actb, 1856LL * 256);
        gemm(stream, actb, wQkv + (size_t)i * 768 * 256, gemmC,
             qkv_b + i * 768, 768, 1856, 768, 256, 1, 0, 0, 0);
        k_qkv_split EL(16 * 480 * 64)(gemmC, Qb, Kb, Vt);
        // scores = Q K^T  (16 batches of 480x480x64)
        gemm(stream, Qb, Kb, gemmC, nullptr, 0, 480, 480, 64,
             16, 480LL * 64, 480LL * 64, 480LL * 480);
        k_softmax_bias<<<dim3(480, 16, 1), 256, 0, stream>>>(
            gemmC, rel_emb + (size_t)i * 1025 * 4, imcol);
        // o = attn V  (16 batches of 480x64x480)
        gemm(stream, imcol, Vt, oav, nullptr, 0, 480, 64, 480,
             16, 480LL * 480, 64LL * 480, 480LL * 64);
        k_merge_ctx EL(4 * 464 * 256)(oav, actb);
        gemm(stream, actb, wOutW + (size_t)i * 256 * 256, proj,
             out_b + i * 256, 256, 1856, 256, 256, 1, 0, 0, 0);
        k_add_ln<<<1856, 256, 0, stream>>>(hbuf, proj, ln1_g + i * 256, ln1_b + i * 256);
        // MLP
        k_convert EL(1856LL * 256)(hbuf, actb, 1856LL * 256);
        gemm(stream, actb, wM1 + (size_t)i * 1024 * 256, gemmC,
             mlp_b1 + i * 1024, 1024, 1856, 1024, 256, 1, 0, 0, 0);
        k_gelu_ip EL(1856LL * 1024)(gemmC, 1856LL * 1024);
        k_convert EL(1856LL * 1024)(gemmC, actb, 1856LL * 1024);
        gemm(stream, actb, wM2 + (size_t)i * 256 * 1024, proj,
             mlp_b2 + i * 256, 256, 1856, 256, 1024, 1, 0, 0, 0);
        k_add_ln<<<1856, 256, 0, stream>>>(hbuf, proj, ln2_g + i * 256, ln2_b + i * 256);
    }

    // ---- classifier (N padded 100 -> 128) ----
    k_convert EL(1856LL * 256)(hbuf, actb, 1856LL * 256);
    gemm(stream, actb, wCls, gemmC, cls_b, 100, 1856, 128, 256, 1, 0, 0, 0);
    k_extract EL(1856 * 100)(gemmC, out);

#undef EL
}
